// GRUDecoder_34205119545371
// MI455X (gfx1250) — compile-verified, matrix operand order checked by reading, not run on hardware
//
#include <hip/hip_runtime.h>

typedef __attribute__((ext_vector_type(16))) __bf16        v16bf;
typedef __attribute__((ext_vector_type(8)))  float         v8f;
typedef __attribute__((ext_vector_type(4)))  unsigned int  u32x4;

union Frag { v16bf v; u32x4 u[2]; };

__device__ __forceinline__ unsigned short f2bf(float f) {
  union { float f; unsigned int u; } a; a.f = f;
  unsigned int r = a.u + 0x7FFFu + ((a.u >> 16) & 1u);   // round-to-nearest-even
  return (unsigned short)(r >> 16);
}

// ---------------------------------------------------------------- convert ---
__global__ void cvt_f32_bf16(const float* __restrict__ in,
                             unsigned short* __restrict__ out, size_t n) {
  size_t i = (size_t)blockIdx.x * blockDim.x + threadIdx.x;
  size_t stride = (size_t)gridDim.x * blockDim.x;
  for (; i < n; i += stride) out[i] = f2bf(in[i]);
}

// ------------------------------------------------------------------- GEMM ---
// out[M,N] = act(A[M,K](bf16, row-major) @ W[N,K](bf16, row-major)^T + bias)
// 256 threads = 8 waves; block tile 128x128; wave tile 64x32 (4x2 WMMA tiles).
template <int RELU, int OUTBF16>
__global__ __launch_bounds__(256) void wmma_gemm(
    const __bf16* __restrict__ A, const __bf16* __restrict__ W,
    const float* __restrict__ bias, void* __restrict__ out,
    int M, int N, int K) {
  const int lane  = threadIdx.x & 31;
  const int wave  = threadIdx.x >> 5;
  const int half  = lane >> 4;       // selects K sub-range per 16-bit layout
  const int l15   = lane & 15;
  const int waveM = wave >> 2;       // 0..1 -> 64 rows
  const int waveN = wave & 3;        // 0..3 -> 32 cols
  const int m0 = blockIdx.x * 128 + waveM * 64;
  const int n0 = blockIdx.y * 128 + waveN * 32;
  (void)M;

  v8f acc[4][2] = {};

  for (int k0 = 0; k0 < K; k0 += 32) {
    Frag a[4];
#pragma unroll
    for (int i = 0; i < 4; ++i) {
      const __bf16* p = A + (size_t)(m0 + i * 16 + l15) * K + k0 + half * 8;
      a[i].u[0] = *(const u32x4*)p;         // K = k0+half*8 .. +7
      a[i].u[1] = *(const u32x4*)(p + 16);  // K = k0+half*8+16 .. +23
    }
    Frag b[2];
#pragma unroll
    for (int j = 0; j < 2; ++j) {
      int n = n0 + j * 16 + l15;
      if (n < N) {
        const __bf16* p = W + (size_t)n * K + k0 + half * 8;
        b[j].u[0] = *(const u32x4*)p;
        b[j].u[1] = *(const u32x4*)(p + 16);
      } else {
        b[j].u[0] = (u32x4)0u;
        b[j].u[1] = (u32x4)0u;
      }
    }
#pragma unroll
    for (int i = 0; i < 4; ++i)
#pragma unroll
      for (int j = 0; j < 2; ++j)
        acc[i][j] = __builtin_amdgcn_wmma_f32_16x16x32_bf16(
            false, a[i].v, false, b[j].v, (short)0, acc[i][j], false, false);
  }

#pragma unroll
  for (int j = 0; j < 2; ++j) {
    int n = n0 + j * 16 + l15;
    float bv = (n < N) ? bias[n] : 0.f;
#pragma unroll
    for (int i = 0; i < 4; ++i) {
#pragma unroll
      for (int r = 0; r < 8; ++r) {
        int m = m0 + i * 16 + half * 8 + r;   // C/D layout: lane-half selects M+8
        float c = acc[i][j][r] + bv;
        if (RELU) c = fmaxf(c, 0.f);
        if (n < N) {
          if (OUTBF16)
            ((unsigned short*)out)[(size_t)m * N + n] = f2bf(c);
          else
            ((float*)out)[(size_t)m * N + n] = c;
        }
      }
    }
  }
}

// ---------------------------------------------------------- GRU recurrence ---
// Persistent: 32 blocks x 128 threads (4 waves). Block owns 32 gate-cols in
// EACH of the r/z/n gate blocks; its 96 W_hh rows (3x32 x 1024 K, 192 KB) are
// staged once into LDS (CDNA5 WGP has 320 KB) and re-read every timestep via
// ds_load_b128 -- weights never touch global/scratch inside the scan loop.
// Wave owns one (m,n) 16x16 tile position replicated across the 3 gates, so
// the sigmoid/tanh/blend is register-local. Hidden state fp32 lives in VGPRs;
// the bf16 copy ping-pongs through two global buffers with a per-step grid
// barrier (arrive counter + s_sleep spin).
__global__ __launch_bounds__(128, 1) void gru_scan(
    const __bf16* __restrict__ Whh, const float* __restrict__ bhh,
    const float* __restrict__ xg, __bf16* __restrict__ hbuf0,
    __bf16* __restrict__ hbuf1, __bf16* __restrict__ hs,
    unsigned int* __restrict__ bar, int T) {
  const int H = 1024;
  extern __shared__ __align__(16) unsigned short ldsW[];  // 96 rows x 1024 bf16

  const int tid  = threadIdx.x;
  const int lane = tid & 31;
  const int wave = tid >> 5;           // 0..3
  const int half = lane >> 4;
  const int l15  = lane & 15;
  const int mt   = wave >> 1;          // 0..1 (batch rows 0-15 / 16-31)
  const int nt   = wave & 1;           // 0..1 (16-col slice)
  const int colb = blockIdx.x * 32;    // block's gate-local column base
  const int nloc = nt * 16 + l15;      // 0..31 within slice
  const int ncol = colb + nloc;        // gate-local column
  const int arow = mt * 16 + l15;      // A fragment row (batch)

  // ---- stage W_hh slice into LDS: local row (g*32+c) <- global row g*H+colb+c
  for (int v = tid; v < 96 * 128; v += blockDim.x) {   // 128 x 16B per row
    int row = v >> 7;
    int cv  = v & 127;
    int g   = row >> 5;
    int c   = row & 31;
    const u32x4* src =
        (const u32x4*)(Whh + (size_t)(g * H + colb + c) * H) + cv;
    ((u32x4*)&ldsW[(size_t)row * H])[cv] = *src;
  }
  __syncthreads();

  const float bR = bhh[ncol];
  const float bZ = bhh[H + ncol];
  const float bN = bhh[2 * H + ncol];

  float h[8];
#pragma unroll
  for (int r = 0; r < 8; ++r) h[r] = 0.f;

  for (int t = 0; t < T; ++t) {
    const __bf16*   hin  = (t & 1) ? hbuf1 : hbuf0;
    unsigned short* hout = (unsigned short*)((t & 1) ? hbuf0 : hbuf1);

    v8f aR = {}, aZ = {}, aN = {};
    for (int k0 = 0; k0 < H; k0 += 32) {
      Frag a, wR, wZ, wN;
      const __bf16* ap = hin + (size_t)arow * H + k0 + half * 8;
      a.u[0] = *(const u32x4*)ap;
      a.u[1] = *(const u32x4*)(ap + 16);
      const unsigned short* r0 = &ldsW[(size_t)(0 * 32 + nloc) * H + k0 + half * 8];
      const unsigned short* r1 = &ldsW[(size_t)(1 * 32 + nloc) * H + k0 + half * 8];
      const unsigned short* r2 = &ldsW[(size_t)(2 * 32 + nloc) * H + k0 + half * 8];
      wR.u[0] = *(const u32x4*)r0;  wR.u[1] = *(const u32x4*)(r0 + 16);
      wZ.u[0] = *(const u32x4*)r1;  wZ.u[1] = *(const u32x4*)(r1 + 16);
      wN.u[0] = *(const u32x4*)r2;  wN.u[1] = *(const u32x4*)(r2 + 16);
      aR = __builtin_amdgcn_wmma_f32_16x16x32_bf16(false, a.v, false, wR.v,
                                                   (short)0, aR, false, false);
      aZ = __builtin_amdgcn_wmma_f32_16x16x32_bf16(false, a.v, false, wZ.v,
                                                   (short)0, aZ, false, false);
      aN = __builtin_amdgcn_wmma_f32_16x16x32_bf16(false, a.v, false, wN.v,
                                                   (short)0, aN, false, false);
    }

#pragma unroll
    for (int r = 0; r < 8; ++r) {
      int b = mt * 16 + half * 8 + r;                 // batch index
      size_t xoff = ((size_t)b * T + t) * (size_t)(3 * H);
      float xr = xg[xoff + ncol];
      float xz = xg[xoff + H + ncol];
      float xn = xg[xoff + 2 * H + ncol];
      if (t + 1 < T) {  // warm L2/L0 for next step's gate pre-activations
        __builtin_prefetch(&xg[xoff + 3 * H + ncol], 0, 0);
      }
      float hr = aR[r] + bR;
      float hz = aZ[r] + bZ;
      float hn = aN[r] + bN;
      float rg = 1.f / (1.f + __expf(-(xr + hr)));
      float zg = 1.f / (1.f + __expf(-(xz + hz)));
      float ng = tanhf(xn + rg * hn);
      float hnew = (1.f - zg) * ng + zg * h[r];
      h[r] = hnew;
      unsigned short hb = f2bf(hnew);
      hout[(size_t)b * H + ncol] = hb;
      ((unsigned short*)hs)[((size_t)b * T + t) * H + ncol] = hb;
    }

    // grid-wide barrier for step t (per-step counter, zeroed each launch)
    __threadfence();
    __syncthreads();
    if (threadIdx.x == 0) {
      atomicAdd(&bar[t], 1u);
      while (atomicAdd(&bar[t], 0u) < gridDim.x) {
        __builtin_amdgcn_s_sleep(1);
      }
    }
    __syncthreads();
    __threadfence();
  }
}

// ------------------------------------------------------------------ launch ---
extern "C" void kernel_launch(void* const* d_in, const int* in_sizes, int n_in,
                              void* d_out, int out_size, void* d_ws,
                              size_t ws_size, hipStream_t stream) {
  (void)in_sizes; (void)n_in; (void)out_size; (void)ws_size;
  const float* x    = (const float*)d_in[0];
  const float* W1   = (const float*)d_in[1];
  const float* b1   = (const float*)d_in[2];
  const float* W2   = (const float*)d_in[3];
  const float* b2   = (const float*)d_in[4];
  const float* Wih  = (const float*)d_in[5];
  const float* bih  = (const float*)d_in[6];
  const float* Whh  = (const float*)d_in[7];
  const float* bhh  = (const float*)d_in[8];
  const float* Wout = (const float*)d_in[9];
  const float* bout = (const float*)d_in[10];

  const int B = 32, T = 512, I = 512, H = 1024, O = 360;
  const int M = B * T;

  char* ws = (char*)d_ws;
  size_t off = 0;
  auto alloc = [&](size_t bytes) -> void* {
    void* p = ws + off;
    off = (off + bytes + 255) & ~(size_t)255;
    return p;
  };
  unsigned short* w1b   = (unsigned short*)alloc((size_t)H * I * 2);
  unsigned short* w2b   = (unsigned short*)alloc((size_t)H * H * 2);
  unsigned short* wihb  = (unsigned short*)alloc((size_t)3 * H * H * 2);
  unsigned short* whhb  = (unsigned short*)alloc((size_t)3 * H * H * 2);
  unsigned short* woutb = (unsigned short*)alloc((size_t)O * H * 2);
  float*          xgbuf = (float*)alloc((size_t)M * 3 * H * 4);
  unsigned short* xb    = (unsigned short*)xgbuf;  // alias: dead before xg written
  unsigned short* h1b   = (unsigned short*)alloc((size_t)M * H * 2);  // h1, then hs
  unsigned short* h2b   = (unsigned short*)alloc((size_t)M * H * 2);
  unsigned short* hbuf0 = (unsigned short*)alloc((size_t)B * H * 2);
  unsigned short* hbuf1 = (unsigned short*)alloc((size_t)B * H * 2);
  unsigned int*   bar   = (unsigned int*)alloc((size_t)T * 4);

  hipMemsetAsync(hbuf0, 0, (size_t)B * H * 2, stream);  // h0 = 0
  hipMemsetAsync(hbuf1, 0, (size_t)B * H * 2, stream);
  hipMemsetAsync(bar,   0, (size_t)T * 4, stream);

  cvt_f32_bf16<<<1024, 256, 0, stream>>>(x,    xb,    (size_t)M * I);
  cvt_f32_bf16<<<1024, 256, 0, stream>>>(W1,   w1b,   (size_t)H * I);
  cvt_f32_bf16<<<1024, 256, 0, stream>>>(W2,   w2b,   (size_t)H * H);
  cvt_f32_bf16<<<1024, 256, 0, stream>>>(Wih,  wihb,  (size_t)3 * H * H);
  cvt_f32_bf16<<<1024, 256, 0, stream>>>(Whh,  whhb,  (size_t)3 * H * H);
  cvt_f32_bf16<<<1024, 256, 0, stream>>>(Wout, woutb, (size_t)O * H);

  dim3 blk(256);
  // h1 = relu(x @ W1^T + b1)
  wmma_gemm<1, 1><<<dim3(M / 128, H / 128), blk, 0, stream>>>(
      (const __bf16*)xb, (const __bf16*)w1b, b1, h1b, M, H, I);
  // h2 = relu(h1 @ W2^T + b2)
  wmma_gemm<1, 1><<<dim3(M / 128, H / 128), blk, 0, stream>>>(
      (const __bf16*)h1b, (const __bf16*)w2b, b2, h2b, M, H, H);
  // xg = h2 @ W_ih^T + b_ih   (fp32 for recurrence accuracy)
  wmma_gemm<0, 0><<<dim3(M / 128, (3 * H) / 128), blk, 0, stream>>>(
      (const __bf16*)h2b, (const __bf16*)wihb, bih, xgbuf, M, 3 * H, H);
  // GRU scan: persistent, W_hh slice in 192 KB LDS, per-step grid barrier
  gru_scan<<<dim3(32), dim3(128), 96 * 1024 * sizeof(unsigned short), stream>>>(
      (const __bf16*)whhb, bhh, xgbuf, (__bf16*)hbuf0, (__bf16*)hbuf1,
      (__bf16*)h1b, bar, T);
  // out = hs @ W_out^T + b_out  (N=360 handled by column guards)
  wmma_gemm<0, 0><<<dim3(M / 128, (O + 127) / 128), blk, 0, stream>>>(
      (const __bf16*)h1b, (const __bf16*)woutb, bout, d_out, M, O, H);
}
